// ExpertsFeedForward_15436112461976
// MI455X (gfx1250) — matrix operand order repaired and checked by reading, not compiled, not run on hardware
//
#include <hip/hip_runtime.h>

// ---------------- problem constants ----------------
#define D_DIM 512
#define H_DIM 2048
#define E_NUM 16
#define T_TOK 65536
#define CAPN  4096
#define MT    64          // token rows per FFN block
#define TKS   520         // padded row stride (bf16 elems) for D=512 tiles
#define W2S   40          // padded row stride (bf16 elems) for K=32 tiles
#define NCH   (H_DIM / 32)

typedef __bf16 bf16_t;
typedef __attribute__((ext_vector_type(16))) __bf16 v16bf;
typedef __attribute__((ext_vector_type(8)))  __bf16 v8bf;
typedef __attribute__((ext_vector_type(8)))  float  v8f;

union V16 { v16bf v; v8bf h[2]; };

__device__ __forceinline__ float gelu_tanh(float v) {
    float u = 0.7978845608028654f * (v + 0.044715f * v * v * v);
    return 0.5f * v * (1.0f + tanhf(u));
}

// async global -> LDS copy, 16B per lane (GVS mode: s[base] + v_offset)
__device__ __forceinline__ void async_b128(unsigned lds_off, const void* sbase,
                                           unsigned voff) {
    asm volatile("global_load_async_to_lds_b128 %0, %1, %2"
                 :: "v"(lds_off), "v"(voff), "s"(sbase) : "memory");
}
__device__ __forceinline__ void wait_async0() {
    asm volatile("s_wait_asynccnt 0x0" ::: "memory");
}
__device__ __forceinline__ void sched_fence() {
#if __has_builtin(__builtin_amdgcn_sched_barrier)
    __builtin_amdgcn_sched_barrier(0);
#endif
}

// ---------------- 1) router: logits + softmax -> probsT (E,T) ----------------
__global__ __launch_bounds__(256) void router_kernel(
    const float* __restrict__ x, const float* __restrict__ gw,
    float* __restrict__ probsT)
{
    __shared__ float g[D_DIM * E_NUM];            // 32KB
    for (int i = threadIdx.x; i < D_DIM * E_NUM; i += 256) g[i] = gw[i];
    __syncthreads();
    const int t = blockIdx.x * 256 + threadIdx.x;
    const float* xr = x + (size_t)t * D_DIM;
    float acc[E_NUM];
#pragma unroll
    for (int e = 0; e < E_NUM; ++e) acc[e] = 0.f;
    for (int d = 0; d < D_DIM; ++d) {
        const float xv = xr[d];
        const float* gr = &g[d * E_NUM];
#pragma unroll
        for (int e = 0; e < E_NUM; ++e) acc[e] = fmaf(xv, gr[e], acc[e]);
    }
    float mx = acc[0];
#pragma unroll
    for (int e = 1; e < E_NUM; ++e) mx = fmaxf(mx, acc[e]);
    float sum = 0.f;
#pragma unroll
    for (int e = 0; e < E_NUM; ++e) { acc[e] = __expf(acc[e] - mx); sum += acc[e]; }
    const float inv = 1.f / sum;
#pragma unroll
    for (int e = 0; e < E_NUM; ++e)
        probsT[(size_t)e * T_TOK + t] = acc[e] * inv;
}

// ---------------- 2) radix-select top-CAPN per expert ----------------
__global__ __launch_bounds__(256) void hist_hi_kernel(
    const float* __restrict__ probsT, unsigned* __restrict__ hist)
{
    const int idx = blockIdx.x * 256 + threadIdx.x;      // over E*T
    const int e = idx >> 16;                              // T == 65536
    const unsigned k = __float_as_uint(probsT[idx]);
    atomicAdd(&hist[((size_t)e << 16) + (k >> 16)], 1u);
}

__global__ __launch_bounds__(256) void hist_lo_kernel(
    const float* __restrict__ probsT, const unsigned* __restrict__ bin_hi,
    unsigned* __restrict__ hist)
{
    const int idx = blockIdx.x * 256 + threadIdx.x;
    const int e = idx >> 16;
    const unsigned k = __float_as_uint(probsT[idx]);
    if ((k >> 16) == bin_hi[e])
        atomicAdd(&hist[((size_t)e << 16) + (k & 0xffffu)], 1u);
}

// per-expert: scan 65536 bins from the top, find bin where cumulative >= cap
__global__ __launch_bounds__(1024) void find_thresh_kernel(
    const unsigned* __restrict__ hist, const int* __restrict__ prev_above,
    unsigned* __restrict__ bin_out, int* __restrict__ above_out)
{
    const int e = blockIdx.x;
    const int cap = prev_above ? (CAPN - prev_above[e]) : CAPN;
    __shared__ unsigned s[1024];
    __shared__ int s_min;
    const int tid = threadIdx.x;
    if (tid == 0) s_min = 1 << 30;
    __syncthreads();
    int running = 0;
    for (int cb = 63; cb >= 0; --cb) {
        const int bin = cb * 1024 + 1023 - tid;          // descending with tid
        const unsigned c = hist[((size_t)e << 16) + bin];
        s[tid] = c;
        __syncthreads();
        for (int off = 1; off < 1024; off <<= 1) {       // inclusive scan
            unsigned add = (tid >= off) ? s[tid - off] : 0u;
            __syncthreads();
            s[tid] += add;
            __syncthreads();
        }
        const unsigned pref = s[tid];
        const unsigned total = s[1023];
        __syncthreads();
        if (running + (int)total >= cap) {               // uniform branch
            if (running + (int)pref >= cap) atomicMin(&s_min, tid);
            __syncthreads();
            if (tid == s_min) {
                bin_out[e] = (unsigned)bin;
                above_out[e] = running + (int)pref - (int)c;
            }
            break;
        }
        running += (int)total;
    }
}

__global__ void finalize_kernel(
    const unsigned* __restrict__ bin_hi, const int* __restrict__ above_hi,
    const unsigned* __restrict__ bin_lo, const int* __restrict__ above_lo,
    unsigned* __restrict__ keyT, int* __restrict__ n_gt, int* __restrict__ need_eq)
{
    const int e = threadIdx.x;
    if (e < E_NUM) {
        keyT[e] = (bin_hi[e] << 16) | bin_lo[e];
        const int g = above_hi[e] + above_lo[e];
        n_gt[e] = g;
        need_eq[e] = CAPN - g;
    }
}

__global__ __launch_bounds__(256) void compact_kernel(
    const float* __restrict__ probsT, const unsigned* __restrict__ keyT,
    const int* __restrict__ n_gt, const int* __restrict__ need_eq,
    int* __restrict__ cnt_gt, int* __restrict__ cnt_eq,
    int* __restrict__ tok, float* __restrict__ sc)
{
    const int idx = blockIdx.x * 256 + threadIdx.x;
    const int e = idx >> 16;
    const int t = idx & 0xffff;
    const float p = probsT[idx];
    const unsigned k = __float_as_uint(p);
    const unsigned K = keyT[e];
    if (k > K) {
        const int s = atomicAdd(&cnt_gt[e], 1);
        tok[(size_t)e * CAPN + s] = t;
        sc[(size_t)e * CAPN + s] = p;
    } else if (k == K) {
        const int q = atomicAdd(&cnt_eq[e], 1);
        if (q < need_eq[e]) {
            const int s = n_gt[e] + q;
            tok[(size_t)e * CAPN + s] = t;
            sc[(size_t)e * CAPN + s] = p;
        }
    }
}

// ---------------- 3) fused FFN (shared & routed), WMMA bf16 + async staging ----
template <bool ROUTED>
__global__ __launch_bounds__(256, 1) void ffn_kernel(
    const float* __restrict__ x,
    const float* __restrict__ w1, const float* __restrict__ b1,
    const float* __restrict__ w2, const float* __restrict__ b2,
    const int* __restrict__ tok_idx, const float* __restrict__ scoresG,
    float* __restrict__ out)
{
    __shared__ bf16_t toks[MT * TKS];        // 64 x 512 token tile (bf16)
    __shared__ bf16_t w1s[32 * TKS];         // [h_local][d] transposed w1 chunk
    __shared__ bf16_t w2s[D_DIM * W2S];      // [d][h_local] transposed w2 chunk
    __shared__ bf16_t hs[MT * W2S];          // 64 x 32 activations (bf16)
    __shared__ float  stage_w1[D_DIM * 32];  // raw fp32 w1 chunk (async dest)
    __shared__ float  stage_w2[32 * D_DIM];  // raw fp32 w2 chunk (async dest)
    __shared__ int    rowtok[MT];
    __shared__ float  rowsc[MT];

    const int tid = threadIdx.x;
    const int e = ROUTED ? blockIdx.y : 0;
    const int tileBase = blockIdx.x * MT;
    const float* w1e = w1 + (ROUTED ? (size_t)e * D_DIM * H_DIM : 0);
    const float* b1e = b1 + (ROUTED ? (size_t)e * H_DIM : 0);
    const float* w2e = w2 + (ROUTED ? (size_t)e * H_DIM * D_DIM : 0);
    const float* b2e = b2 + (ROUTED ? (size_t)e * D_DIM : 0);

    const unsigned stage1_off = (unsigned)(size_t)(void*)stage_w1;
    const unsigned stage2_off = (unsigned)(size_t)(void*)stage_w2;

    // issue async staging for a given H-chunk (32 h columns): 16B per lane
    auto issue_chunk = [&](int hc) {
        const char* p1 = (const char*)(w1e + (size_t)hc * 32);
        const char* p2 = (const char*)(w2e + (size_t)hc * 32 * D_DIM);
#pragma unroll
        for (int it = 0; it < 16; ++it) {
            const int r = it * 256 + tid;                 // 4096 16B runs each
            {   // w1 chunk: [d][h0..31] rows; run = (d, 4 h's)
                const int d = r >> 3, hq = r & 7;
                async_b128(stage1_off + r * 16, p1,
                           (unsigned)(d * (H_DIM * 4) + hq * 16));
            }
            {   // w2 chunk: [h][d0..511] rows; run = (h, 4 d's)
                const int h = r >> 7, dq = r & 127;
                async_b128(stage2_off + r * 16, p2,
                           (unsigned)(h * (D_DIM * 4) + dq * 16));
            }
        }
    };

    if (ROUTED) {
        if (tid < MT) {
            rowtok[tid] = tok_idx[(size_t)e * CAPN + tileBase + tid];
            rowsc[tid]  = scoresG[(size_t)e * CAPN + tileBase + tid];
        }
        __syncthreads();
    }
    issue_chunk(0);                                   // overlap with token load
    for (int i = tid; i < MT * D_DIM; i += 256) {
        const int m = i >> 9, d = i & (D_DIM - 1);
        const int row = ROUTED ? rowtok[m] : (tileBase + m);
        toks[m * TKS + d] = (bf16_t)x[(size_t)row * D_DIM + d];
    }

    const int w  = tid >> 5;
    const int L  = tid & 31;
    const int lh = L >> 4;       // lane half 0/1
    const int ll = L & 15;

    const int m1 = w & 3, hn = w >> 2;            // layer-1: 4x2 tiles of 16x16
    const int m2 = w & 3, nb = (w >> 2) * 256;    // layer-2: wave owns 16 N-tiles

    v8f zero = {};
    v8f acc2[16];
#pragma unroll
    for (int j = 0; j < 16; ++j) acc2[j] = zero;

    for (int hc = 0; hc < NCH; ++hc) {
        wait_async0();            // this wave's async chunk arrived
        __syncthreads();          // => every wave's chunk arrived; prev reads done

        // convert staged fp32 chunk -> bf16 WMMA layouts (transpose via LDS)
        for (int i = tid; i < 32 * D_DIM; i += 256) {
            const int h = i & 31, d = i >> 5;
            w1s[h * TKS + d] = (bf16_t)stage_w1[d * 32 + h];
        }
        for (int i = tid; i < 32 * D_DIM; i += 256) {
            const int d = i & (D_DIM - 1), h = i >> 9;
            w2s[d * W2S + h] = (bf16_t)stage_w2[h * D_DIM + d];
        }
        __syncthreads();          // converts visible; staging free for reuse

        if (hc + 1 < NCH) issue_chunk(hc + 1);    // prefetch next chunk async

        // ---- layer 1: h_tile(16x16) = toks(16x512) @ w1(512x16) ----
        v8f acc1 = zero;
        V16 a[2], b[2];
        auto load1 = [&](int kk, V16& A, V16& B) {
            const int abase = (m1 * 16 + ll) * TKS + kk * 32 + lh * 8;
            A.h[0] = *(const v8bf*)&toks[abase];
            A.h[1] = *(const v8bf*)&toks[abase + 16];
            const int bbase = (hn * 16 + ll) * TKS + kk * 32 + lh * 8;
            B.h[0] = *(const v8bf*)&w1s[bbase];
            B.h[1] = *(const v8bf*)&w1s[bbase + 16];
        };
        load1(0, a[0], b[0]);
        load1(1, a[1], b[1]);
        sched_fence();
#pragma unroll
        for (int kk = 0; kk < 16; ++kk) {         // depth-2 software pipeline
            acc1 = __builtin_amdgcn_wmma_f32_16x16x32_bf16(
                false, a[kk & 1].v, false, b[kk & 1].v, (short)0, acc1,
                false, false);
            if (kk + 2 < 16) load1(kk + 2, a[kk & 1], b[kk & 1]);
            sched_fence();                        // pin wmma(k) | load(k+2)
        }
        {
            const float bias = b1e[hc * 32 + hn * 16 + ll];
#pragma unroll
            for (int r = 0; r < 8; ++r) {
                const int row = m1 * 16 + r + lh * 8;
                hs[row * W2S + hn * 16 + ll] = (bf16_t)gelu_tanh(acc1[r] + bias);
            }
        }
        __syncthreads();

        // ---- layer 2: y(64x512) += hs(64x32) @ w2(32x512) ----
        V16 a2, bb[2];
        auto load2 = [&](int j, V16& B) {
            const int bbx = (nb + j * 16 + ll) * W2S + lh * 8;
            B.h[0] = *(const v8bf*)&w2s[bbx];
            B.h[1] = *(const v8bf*)&w2s[bbx + 16];
        };
        {
            const int a2base = (m2 * 16 + ll) * W2S + lh * 8;
            a2.h[0] = *(const v8bf*)&hs[a2base];
            a2.h[1] = *(const v8bf*)&hs[a2base + 16];
        }
        load2(0, bb[0]);
        load2(1, bb[1]);
        sched_fence();
#pragma unroll
        for (int j = 0; j < 16; ++j) {            // depth-2 software pipeline
            acc2[j] = __builtin_amdgcn_wmma_f32_16x16x32_bf16(
                false, a2.v, false, bb[j & 1].v, (short)0, acc2[j],
                false, false);
            if (j + 2 < 16) load2(j + 2, bb[j & 1]);
            sched_fence();                        // pin wmma(j) | load(j+2)
        }
    }

    // ---- epilogue ----
#pragma unroll
    for (int j = 0; j < 16; ++j) {
        const int col = nb + j * 16 + ll;
        const float bias = b2e[col];
#pragma unroll
        for (int r = 0; r < 8; ++r) {
            const int row = m2 * 16 + r + lh * 8;
            const float v = acc2[j][r] + bias;
            if (ROUTED) {
                unsafeAtomicAdd(&out[(size_t)rowtok[row] * D_DIM + col],
                                v * rowsc[row]);
            } else {
                out[(size_t)(tileBase + row) * D_DIM + col] = v;
            }
        }
    }
}

// ---------------- launch ----------------
extern "C" void kernel_launch(void* const* d_in, const int* in_sizes, int n_in,
                              void* d_out, int out_size, void* d_ws, size_t ws_size,
                              hipStream_t stream) {
    const float* x      = (const float*)d_in[0];
    const float* gate_w = (const float*)d_in[1];
    const float* w1     = (const float*)d_in[2];
    const float* b1     = (const float*)d_in[3];
    const float* w2     = (const float*)d_in[4];
    const float* b2     = (const float*)d_in[5];
    const float* sw1    = (const float*)d_in[6];
    const float* sb1    = (const float*)d_in[7];
    const float* sw2    = (const float*)d_in[8];
    const float* sb2    = (const float*)d_in[9];
    float* out = (float*)d_out;

    char* ws = (char*)d_ws;
    size_t off = 0;
    float* probsT = (float*)(ws + off);        off += (size_t)E_NUM * T_TOK * 4;  // 4MB
    unsigned* hist_hi = (unsigned*)(ws + off); off += (size_t)E_NUM * 65536 * 4;  // 4MB
    unsigned* hist_lo = (unsigned*)(ws + off); off += (size_t)E_NUM * 65536 * 4;  // 4MB
    char* smallblk = ws + off;                 off += 1024;
    unsigned* bin_hi   = (unsigned*)(smallblk + 0);
    int*      above_hi = (int*)(smallblk + 64);
    unsigned* bin_lo   = (unsigned*)(smallblk + 128);
    int*      above_lo = (int*)(smallblk + 192);
    unsigned* keyT     = (unsigned*)(smallblk + 256);
    int*      n_gt     = (int*)(smallblk + 320);
    int*      need_eq  = (int*)(smallblk + 384);
    int*      cnt_gt   = (int*)(smallblk + 448);
    int*      cnt_eq   = (int*)(smallblk + 512);
    int*   d_tok = (int*)(ws + off);           off += (size_t)E_NUM * CAPN * 4;   // 256KB
    float* d_sc  = (float*)(ws + off);         off += (size_t)E_NUM * CAPN * 4;   // 256KB
    (void)ws_size; (void)in_sizes; (void)n_in; (void)out_size;

    // zero histograms + counters (one contiguous region)
    hipMemsetAsync(hist_hi, 0, (size_t)E_NUM * 65536 * 8 + 1024, stream);

    router_kernel<<<T_TOK / 256, 256, 0, stream>>>(x, gate_w, probsT);
    hist_hi_kernel<<<(E_NUM * T_TOK) / 256, 256, 0, stream>>>(probsT, hist_hi);
    find_thresh_kernel<<<E_NUM, 1024, 0, stream>>>(hist_hi, nullptr, bin_hi, above_hi);
    hist_lo_kernel<<<(E_NUM * T_TOK) / 256, 256, 0, stream>>>(probsT, bin_hi, hist_lo);
    find_thresh_kernel<<<E_NUM, 1024, 0, stream>>>(hist_lo, above_hi, bin_lo, above_lo);
    finalize_kernel<<<1, E_NUM, 0, stream>>>(bin_hi, above_hi, bin_lo, above_lo,
                                             keyT, n_gt, need_eq);
    compact_kernel<<<(E_NUM * T_TOK) / 256, 256, 0, stream>>>(
        probsT, keyT, n_gt, need_eq, cnt_gt, cnt_eq, d_tok, d_sc);

    // shared expert initializes out (plain stores), routed experts atomic-add on top
    ffn_kernel<false><<<dim3(T_TOK / MT), 256, 0, stream>>>(
        x, sw1, sb1, sw2, sb2, nullptr, nullptr, out);
    ffn_kernel<true><<<dim3(CAPN / MT, E_NUM), 256, 0, stream>>>(
        x, w1, b1, w2, b2, d_tok, d_sc, out);
}
